// Interaction3_55439437857330
// MI455X (gfx1250) — matrix operand order repaired and checked by reading, not compiled, not run on hardware
//
#include <hip/hip_runtime.h>

typedef __attribute__((ext_vector_type(16))) _Float16 v16h;
typedef __attribute__((ext_vector_type(8)))  _Float16 v8h;
typedef __attribute__((ext_vector_type(8)))  float    v8f;
typedef unsigned int u32x4 __attribute__((ext_vector_type(4)));
typedef int          i32x8 __attribute__((ext_vector_type(8)));
typedef int          i32x4 __attribute__((ext_vector_type(4)));

#define BATCH 2
#define CIN   256
#define HH    96
#define WW    96
#define INTER 128
#define NPIX  (BATCH*HH*WW)   /* 18432 */
#define KC    (CIN*9)         /* 2304  */
#define KSTEPS 72
#define NT    1152            /* pixel tiles of 16 */
#define EPSV  1e-5f

union V16u { v16h v; v8h h[2]; };

static __device__ __forceinline__ v8f wmma16(v16h a, v16h b, v8f c) {
  return __builtin_amdgcn_wmma_f32_16x16x32_f16(false, a, false, b, (short)0, c,
                                                false, false);
}

// ---------------------------------------------------------------------------
// Pack kernel: swizzle conv weights + projection weights into WMMA A-operand
// per-lane layout; fold BN into scale/shift.
// A layout (16-bit, 16x32): lane&15 = row M; basek = (lane&16)?8:0;
// half j -> K = basek + (j<8 ? j : j+8).
// ---------------------------------------------------------------------------
__global__ __launch_bounds__(256) void pack_kernel(
    const float* w1, const float* w2,
    const float* g1, const float* be1, const float* me1, const float* va1,
    const float* g2, const float* be2, const float* me2, const float* va2,
    const float* wq1, const float* wk1, const float* wv1,
    const float* wq2, const float* wk2, const float* wv2,
    _Float16* Apk, _Float16* Wpk, float* bnsc, float* bnsh)
{
  int idx = blockIdx.x * 256 + threadIdx.x;
  const int CP = 2 * 8 * KSTEPS * 512;   // 589824 conv-weight halves
  const int PP = 2 * 3 * 8 * 4 * 512;    // 98304 proj-weight halves
  if (idx < CP) {
    int j = idx & 15, lane = (idx >> 4) & 31, ks = (idx >> 9) % KSTEPS;
    int rest = idx / (512 * KSTEPS);
    int mt = rest & 7, s = rest >> 3;
    int basek = (lane & 16) ? 8 : 0;
    int kg = ks * 32 + (j < 8 ? basek + j : basek + 8 + j);
    int o = mt * 16 + (lane & 15);
    const float* w = s ? w2 : w1;
    Apk[idx] = (_Float16)w[o * KC + kg];      // [O][C][3][3] flat == o*2304+kg
  } else if (idx < CP + PP) {
    int t = idx - CP;
    int j = t & 15, lane = (t >> 4) & 31, ks = (t >> 9) & 3, hd = (t >> 11) & 7;
    int p = (t >> 14) % 3, s = t / 49152;
    int basek = (lane & 16) ? 8 : 0;
    int kch = ks * 32 + (j < 8 ? basek + j : basek + 8 + j);
    int m = lane & 15;
    float val = 0.f;
    if (p == 0) {                       // q: 8 valid rows per head, rest zero
      if (m < 8) { const float* wq = s ? wq2 : wq1; val = wq[(hd*8+m)*INTER + kch]; }
    } else if (p == 1) {
      const float* wk = s ? wk2 : wk1; val = wk[(hd*16+m)*INTER + kch];
    } else {
      const float* wv = s ? wv2 : wv1; val = wv[(hd*16+m)*INTER + kch];
    }
    Wpk[t] = (_Float16)val;
  } else if (idx < CP + PP + 256) {
    int t = idx - CP - PP;
    int s = t >> 7, ch = t & 127;
    const float* g  = s ? g2  : g1;  const float* be = s ? be2 : be1;
    const float* me = s ? me2 : me1; const float* va = s ? va2 : va1;
    float sc = g[ch] * rsqrtf(va[ch] + EPSV);
    bnsc[t] = sc;
    bnsh[t] = be[ch] - me[ch] * sc;
  }
}

// ---------------------------------------------------------------------------
// Conv3x3 + BN + ReLU as implicit GEMM: out[128, 18432] = W[128,2304] * im2col.
// One block per (stream, 16-pixel tile): stage the 2304x16 im2col tile once
// into LDS (72 KB), then 8 waves each compute one 16-row output tile.
// Inner loop per k-step: 2x global b128 (packed A) + 2x ds b128 (B) + 1 wmma.
// ---------------------------------------------------------------------------
__global__ __launch_bounds__(256) void conv_bn_relu_kernel(
    const float* x1, const float* x2, const _Float16* Apk,
    const float* bnsc, const float* bnsh, _Float16* hbuf)
{
  __shared__ _Float16 sIm[KC * 16];          // [px=16][k=2304] f16, 72 KB

  int tid  = threadIdx.x;
  int lane = tid & 31;
  int mt   = tid >> 5;                       // wave id == M tile
  int blk  = blockIdx.x;                     // 0..2303
  int nt   = blk % NT;
  int s    = blk / NT;
  int n0   = nt * 16;
  int b    = n0 / (HH * WW);
  int rem  = n0 - b * HH * WW;
  int y    = rem / WW;
  int x0   = rem - y * WW;                   // 16 px in one row (96%16==0)
  const float* xs = s ? x2 : x1;

  // ---- stage im2col tile (coalesced; bounds handled once here) ------------
  for (int it = 0; it < (KC * 16) / 256; ++it) {   // 144 iterations
    int idx = tid + it * 256;
    int k   = idx >> 4;
    int px  = idx & 15;
    int c   = k / 9;
    int off = k - c * 9;
    int ky  = off / 3;
    int kx  = off - ky * 3;
    int yy  = y + ky - 1;
    int xx  = x0 + px + kx - 1;
    float v = 0.f;
    if (yy >= 0 && yy < HH && xx >= 0 && xx < WW)
      v = xs[((size_t)(b * CIN + c) * HH + yy) * WW + xx];
    sIm[px * KC + k] = (_Float16)v;
  }
  __syncthreads();

  // ---- WMMA main loop ------------------------------------------------------
  const _Float16* ap = Apk + (size_t)((s * 8 + mt) * KSTEPS) * 512 + lane * 16;
  v8f acc = {};
  for (int ks = 0; ks < KSTEPS; ++ks) {
    V16u au;
    au.h[0] = *(const v8h*)ap;
    au.h[1] = *(const v8h*)(ap + 8);
    ap += 512;
    const _Float16* bp = &sIm[(lane & 15) * KC + ks * 32 + ((lane & 16) ? 8 : 0)];   // B: lane = n col
    V16u bu;
    bu.h[0] = *(const v8h*)bp;
    bu.h[1] = *(const v8h*)(bp + 16);
    acc = wmma16(au.v, bu.v, acc);
  }

  int half8 = (lane & 16) ? 8 : 0;
  int n = n0 + (lane & 15);
  #pragma unroll
  for (int r = 0; r < 8; ++r) {
    int orow = mt * 16 + r + half8;          // C/D layout: lanes16-31 -> M=r+8
    float v = acc[r] * bnsc[s * 128 + orow] + bnsh[s * 128 + orow];
    v = fmaxf(v, 0.f);
    hbuf[(size_t)(s * 128 + orow) * NPIX + n] = (_Float16)v;
  }
}

// ---------------------------------------------------------------------------
// Fused window attention: one block per window (288), one wave per head (8).
// Phase0: ONE Tensor-Data-Mover load pulls the whole h1/h2 window tile
//         (x=8 f16, y=8 rows stride 96, z=256 ch-rows stride 18432) into LDS
//         in [ch][iy][ix] order -- exactly the layout the WMMA B operand needs.
// Phase1: q/k/v projections via WMMA. Phase2: S=|QK1'-QK2'|, softmax
// (shfl butterflies over the 16-lane C/D row halves). Phase3: O=attn*V via
// WMMA, normalize, scatter to unwindowed NCHW output.
// ---------------------------------------------------------------------------
__global__ __launch_bounds__(256) void window_attn_kernel(
    const _Float16* hbuf, const _Float16* Wpk,
    const float* bq1, const float* bk1, const float* bv1,
    const float* bq2, const float* bk2, const float* bv2,
    float* out)
{
  __shared__ _Float16 sPool[32768];        // ph0/1: h [2][128][64]; ph2/3: attn [8][64][64]
  __shared__ _Float16 sQ[8][64 * 16];      // qcat [px][dcat]
  __shared__ _Float16 sK[8][2][16 * 64];   // k    [px][d]
  __shared__ _Float16 sV[8][2][64 * 16];   // vT   [d][px]

  int tid  = threadIdx.x;
  int lane = tid & 31;
  int hd   = tid >> 5;
  int wid  = blockIdx.x;
  int b  = wid / 144;
  int wy = (wid / 12) % 12;
  int wx = wid % 12;
  int ybase = wy * 8, xbase = wx * 8;
  int half8 = (lane & 16) ? 8 : 0;

  // ---- phase 0: TDM tensor load, h tiles -> LDS as [s][ch][px] ------------
  if (hd == 0) {
    size_t nbase = (size_t)b * HH * WW + (size_t)ybase * WW + xbase;
    unsigned long long gaddr = (unsigned long long)(size_t)(hbuf + nbase);
    unsigned int laddr = (unsigned int)(size_t)(&sPool[0]);
    u32x4 g0;
    g0[0] = 1u;                                            // count=1 (valid D#)
    g0[1] = laddr;                                         // lds_addr
    g0[2] = (unsigned int)(gaddr & 0xffffffffull);         // global_addr lo
    g0[3] = (unsigned int)((gaddr >> 32) & 0x1ffffffull)   // global_addr hi
          | (2u << 30);                                    // type=2 (image)
    i32x8 g1;
    g1[0] = 0x10000;                   // data_size=1 (2 bytes); wg_mask=0
    g1[1] = (int)(8u << 16);           // tensor_dim0 = 8   (bits 79:48)
    g1[2] = (int)(8u << 16);           // tensor_dim1 = 8   (bits 111:80)
    g1[3] = (int)(8u << 16);           // tile_dim0   = 8   (bits 127:112)
    g1[4] = (int)((256u << 16) | 8u);  // tile_dim2=256 | tile_dim1=8
    g1[5] = 96;                        // tensor_dim0_stride = 96 (bits 207:160)
    g1[6] = (int)(0x4800u << 16);      // tensor_dim1_stride = 18432 (bits 255:208)
    g1[7] = 0;
    i32x4 g2;
    g2[0] = 256;                       // tensor_dim2 = 256
    g2[1] = 0; g2[2] = 0; g2[3] = 0;   // dim3 / dim2_stride / tile_dim3 unused
    i32x4 g3 = {};
    i32x8 gx = {};                     // trailing group (6-arg builtin variant)
    __builtin_amdgcn_tensor_load_to_lds(g0, g1, g2, g3, gx, 0);
    __builtin_amdgcn_s_wait_tensorcnt(0);
  }
  __syncthreads();

  // ---- phase 1: per-head projections (q 64x8, k 64x16, v 64x16 per stream)
  for (int s = 0; s < 2; ++s) {
    const float* bq = s ? bq2 : bq1;
    const float* bk = s ? bk2 : bk1;
    const float* bv = s ? bv2 : bv1;
    for (int p = 0; p < 3; ++p) {
      const _Float16* wp = Wpk + (size_t)(((s * 3 + p) * 8 + hd) * 4) * 512;
      for (int nt = 0; nt < 4; ++nt) {
        v8f acc = {};
        #pragma unroll
        for (int ks = 0; ks < 4; ++ks) {
          const _Float16* a_p = wp + ks * 512 + lane * 16;
          V16u au; au.h[0] = *(const v8h*)a_p; au.h[1] = *(const v8h*)(a_p + 8);
          V16u bu;
          #pragma unroll
          for (int i = 0; i < 16; ++i) {
            int ch = ks * 32 + (i < 8 ? half8 + i : 16 + half8 + (i - 8));
            bu.v[i] = sPool[(s * 128 + ch) * 64 + nt * 16 + (lane & 15)];
          }
          acc = wmma16(au.v, bu.v, acc);
        }
        int px = nt * 16 + (lane & 15);
        #pragma unroll
        for (int r = 0; r < 8; ++r) {
          int m = r + half8;
          float v = acc[r];
          if (p == 0) {
            if (m < 8) sQ[hd][px * 16 + s * 8 + m] = (_Float16)(v + bq[hd * 8 + m]);
          } else if (p == 1) {
            sK[hd][s][px * 16 + m] = (_Float16)(v + bk[hd * 16 + m]);
          } else {
            sV[hd][s][m * 64 + px] = (_Float16)(v + bv[hd * 16 + m]);
          }
        }
      }
    }
  }
  __syncthreads();

  // ---- phase 2: scores + softmax ------------------------------------------
  float l[4][8];
  for (int qt = 0; qt < 4; ++qt) {
    int qa = qt * 16 + (lane & 15);
    const _Float16* qp = &sQ[hd][qa * 16 + half8];
    v8h z = {};
    V16u aq; aq.h[0] = *(const v8h*)qp; aq.h[1] = z;   // K(d)=16..31 zero-pad
    float sab[4][8];
    for (int kt = 0; kt < 4; ++kt) {
      V16u b1, b2;
      {                                                // B: lane = n(key), d = half8+s
        const _Float16* k1p = &sK[hd][0][(kt * 16 + (lane & 15)) * 16 + half8];
        const _Float16* k2p = &sK[hd][1][(kt * 16 + (lane & 15)) * 16 + half8];
        b1.h[0] = *(const v8h*)k1p; b1.h[1] = z;
        b2.h[0] = *(const v8h*)k2p; b2.h[1] = z;
      }
      v8f c0 = {};
      v8f s1 = wmma16(aq.v, b1.v, c0);
      v8f s2 = wmma16(aq.v, b2.v, c0);
      #pragma unroll
      for (int r = 0; r < 8; ++r) sab[kt][r] = fabsf(s1[r] - s2[r]);
    }
    #pragma unroll
    for (int r = 0; r < 8; ++r) {
      float m = fmaxf(fmaxf(sab[0][r], sab[1][r]), fmaxf(sab[2][r], sab[3][r]));
      for (int off = 1; off < 16; off <<= 1) m = fmaxf(m, __shfl_xor(m, off));
      float sum = 0.f;
      #pragma unroll
      for (int kt = 0; kt < 4; ++kt) {
        float e = __expf(sab[kt][r] - m); sab[kt][r] = e; sum += e;
      }
      for (int off = 1; off < 16; off <<= 1) sum += __shfl_xor(sum, off);
      l[qt][r] = sum;
      int q = qt * 16 + r + half8;                     // C/D row mapping
      #pragma unroll
      for (int kt = 0; kt < 4; ++kt)
        sPool[hd * 4096 + q * 64 + kt * 16 + (lane & 15)] = (_Float16)sab[kt][r];
    }
  }
  __syncthreads();

  // ---- phase 3: O = attn * V, normalize, scatter to output ---------------
  for (int s = 0; s < 2; ++s) {
    for (int qt = 0; qt < 4; ++qt) {
      v8f acc = {};
      #pragma unroll
      for (int ks = 0; ks < 2; ++ks) {
        int qa = qt * 16 + (lane & 15);
        const _Float16* a_p = &sPool[hd * 4096 + qa * 64 + ks * 32 + half8];
        V16u aa; aa.h[0] = *(const v8h*)a_p; aa.h[1] = *(const v8h*)(a_p + 16);
        int key = ks * 32 + half8;                     // B: lane = n(d)
        const _Float16* b_p = &sV[hd][s][(lane & 15) * 64 + key];
        V16u bb; bb.h[0] = *(const v8h*)b_p; bb.h[1] = *(const v8h*)(b_p + 16);
        acc = wmma16(aa.v, bb.v, acc);
      }
      int d = lane & 15;
      int ch = hd * 16 + d;
      #pragma unroll
      for (int r = 0; r < 8; ++r) {
        int q = qt * 16 + r + half8;
        int iy = q >> 3, ix = q & 7;
        float v = acc[r] / l[qt][r];
        out[(size_t)s * 2359296 +
            ((size_t)(b * 128 + ch) * HH + (ybase + iy)) * WW + (xbase + ix)] = v;
      }
    }
  }
}

// ---------------------------------------------------------------------------
extern "C" void kernel_launch(void* const* d_in, const int* in_sizes, int n_in,
                              void* d_out, int out_size, void* d_ws, size_t ws_size,
                              hipStream_t stream) {
  (void)in_sizes; (void)n_in; (void)out_size; (void)ws_size;
  const float* x1  = (const float*)d_in[0];
  const float* x2  = (const float*)d_in[1];
  const float* w1  = (const float*)d_in[2];
  const float* g1  = (const float*)d_in[3];
  const float* be1 = (const float*)d_in[4];
  const float* me1 = (const float*)d_in[5];
  const float* va1 = (const float*)d_in[6];
  const float* w2  = (const float*)d_in[7];
  const float* g2  = (const float*)d_in[8];
  const float* be2 = (const float*)d_in[9];
  const float* me2 = (const float*)d_in[10];
  const float* va2 = (const float*)d_in[11];
  const float* wq1 = (const float*)d_in[12];
  const float* bq1 = (const float*)d_in[13];
  const float* wk1 = (const float*)d_in[14];
  const float* bk1 = (const float*)d_in[15];
  const float* wv1 = (const float*)d_in[16];
  const float* bv1 = (const float*)d_in[17];
  const float* wq2 = (const float*)d_in[18];
  const float* bq2 = (const float*)d_in[19];
  const float* wk2 = (const float*)d_in[20];
  const float* bk2 = (const float*)d_in[21];
  const float* wv2 = (const float*)d_in[22];
  const float* bv2 = (const float*)d_in[23];

  char* ws = (char*)d_ws;
  _Float16* Apk  = (_Float16*)ws;                         // 589824 halves
  _Float16* Wpk  = Apk + 589824;                          // 98304 halves
  float*    bnsc = (float*)(ws + (589824 + 98304) * 2);   // 256 floats
  float*    bnsh = bnsc + 256;                            // 256 floats
  _Float16* hbuf = (_Float16*)(bnsh + 256);               // 2*128*18432 halves

  pack_kernel<<<2689, 256, 0, stream>>>(w1, w2, g1, be1, me1, va1,
                                        g2, be2, me2, va2,
                                        wq1, wk1, wv1, wq2, wk2, wv2,
                                        Apk, Wpk, bnsc, bnsh);
  conv_bn_relu_kernel<<<2304, 256, 0, stream>>>(x1, x2, Apk, bnsc, bnsh, hbuf);
  window_attn_kernel<<<288, 256, 0, stream>>>(hbuf, Wpk,
                                              bq1, bk1, bv1, bq2, bk2, bv2,
                                              (float*)d_out);
}
